// GCN_47682726920576
// MI455X (gfx1250) — matrix operand order repaired and checked by reading, not compiled, not run on hardware
//
#include <hip/hip_runtime.h>
#include <hip/hip_bf16.h>

// ---------------------------------------------------------------------------
// MI455X (gfx1250) GCN forward.  Memory-bound problem (~600MB HBM traffic,
// ~26us floor @ 23.3TB/s; only ~10.6 GFLOP of GEMM).  GEMMs use
// v_wmma_f32_16x16x32_f16 (wave32, 16x16 tile per wave, K stepped by 32),
// weights staged TRANSPOSED in LDS as f16 so each lane's B fragment is two
// 16B ds loads.  Edge aggregation + pooling use global atomics whose
// working set (51MB) lives in the 192MB L2.
// ---------------------------------------------------------------------------

typedef __attribute__((ext_vector_type(16))) _Float16 v16h;
typedef __attribute__((ext_vector_type(8)))  _Float16 v8h;
typedef __attribute__((ext_vector_type(8)))  float    v8f;

#define NNODES 100000
#define NEDGES 600000
#define NGRAPH 64
#define FIN    32
#define WID    128

// ---------------------------------------------------------------------------
// Generic WMMA GEMM: Out[n, M] = A[n, K] @ Wt[K, M] (+ bias) (+ relu)
// One wave -> one 16x16 output tile.  blockDim.x = 32*(M/16).
// REQUIRES nRows % 16 == 0 (true for 100000 and 64) -> no store guards,
// EXEC stays all-ones through the WMMA (ISA requirement).
// ---------------------------------------------------------------------------
template <int K, int M, bool RELU, bool BIAS>
__global__ void wmma_gemm(const float* __restrict__ A,
                          const float* __restrict__ Wt,
                          const float* __restrict__ bias,
                          float* __restrict__ Out) {
    // Transposed weight tile in LDS, padded pitch to spread LDS banks.
    constexpr int LDB = K + 8;
    __shared__ _Float16 sWt[M * LDB];
    for (int i = threadIdx.x; i < K * M; i += blockDim.x) {
        int k = i / M, m = i - k * M;          // coalesced read of Wt[k][m]
        sWt[m * LDB + k] = (_Float16)Wt[i];
    }
    __syncthreads();

    const int lane = threadIdx.x & 31;
    const int wave = threadIdx.x >> 5;
    const int hi   = lane >> 4;            // 0 for lanes 0-15, 1 for 16-31
    const int l16  = lane & 15;

    const int rowBase = blockIdx.x * 16;
    const int row     = rowBase + l16;
    const int col     = wave * 16 + l16;

    v8f c = {};
    for (int k0 = 0; k0 < K; k0 += 32) {
        // A fragment: lanes 0-15 hold K = k0+{0..7,16..23}; lanes 16-31 -> +8.
        const int kbA = k0 + (hi ? 8 : 0);
        const float* ap = A + (size_t)row * K + kbA;
        // Speculative prefetch of the next K-step (gfx1250 global_prefetch;
        // OOB prefetches are silently dropped per ISA 10.5).
        __builtin_prefetch(ap + 32, 0, 1);
        const float4 a0 = *(const float4*)(ap + 0);
        const float4 a1 = *(const float4*)(ap + 4);
        const float4 a2 = *(const float4*)(ap + 16);
        const float4 a3 = *(const float4*)(ap + 20);
        v16h a;
        a[0]  = (_Float16)a0.x; a[1]  = (_Float16)a0.y;
        a[2]  = (_Float16)a0.z; a[3]  = (_Float16)a0.w;
        a[4]  = (_Float16)a1.x; a[5]  = (_Float16)a1.y;
        a[6]  = (_Float16)a1.z; a[7]  = (_Float16)a1.w;
        a[8]  = (_Float16)a2.x; a[9]  = (_Float16)a2.y;
        a[10] = (_Float16)a2.z; a[11] = (_Float16)a2.w;
        a[12] = (_Float16)a3.x; a[13] = (_Float16)a3.y;
        a[14] = (_Float16)a3.z; a[15] = (_Float16)a3.w;

        // B fragment: lanes 0-15 hold K = k0+{0..7,16..23} of column `col`;
        // lanes 16-31 -> +8.  Two 16B LDS loads.
        const int kbB = k0 + (hi ? 8 : 0);
        const _Float16* bp = &sWt[col * LDB + kbB];
        const v8h b0 = *(const v8h*)(bp + 0);
        const v8h b1 = *(const v8h*)(bp + 16);
        v16h b;
#pragma unroll
        for (int i = 0; i < 8; ++i) { b[i] = b0[i]; b[i + 8] = b1[i]; }

        c = __builtin_amdgcn_wmma_f32_16x16x32_f16(
                false, a, false, b, (short)0, c, false, false);
    }

    // C/D layout: VGPR r -> row rowBase+r (lanes 0-15) / rowBase+r+8 (16-31).
    float bv = 0.0f;
    if (BIAS) bv = bias[col];
    float* op = Out + (size_t)(rowBase + (hi ? 8 : 0)) * M + col;
#pragma unroll
    for (int r = 0; r < 8; ++r) {
        float v = c[r] + bv;
        if (RELU) v = fmaxf(v, 0.0f);
        op[(size_t)r * M] = v;
    }
}

// ---------------------------------------------------------------------------
// Elementwise / graph kernels
// ---------------------------------------------------------------------------
__global__ void zero_f(float* p, int n) {
    int i = blockIdx.x * blockDim.x + threadIdx.x;
    if (i < n) p[i] = 0.0f;
}

__global__ void deg_count(const int* __restrict__ dst, int e, float* deg) {
    int i = blockIdx.x * blockDim.x + threadIdx.x;
    if (i < e) atomicAdd(&deg[dst[i]], 1.0f);
}

__global__ void make_dinv(const float* __restrict__ deg, float* dinv, int n) {
    int i = blockIdx.x * blockDim.x + threadIdx.x;
    if (i < n) dinv[i] = rsqrtf(deg[i] + 1.0f);
}

// agg = h * dinv^2  (self-loop term, also initializes the scatter target)
__global__ void self_init(const float* __restrict__ h,
                          const float* __restrict__ dinv,
                          float* __restrict__ agg, int total) {
    int t = blockIdx.x * blockDim.x + threadIdx.x;
    if (t < total) {
        int node = t >> 7;  // /128
        float dv = dinv[node];
        agg[t] = h[t] * dv * dv;
    }
}

// agg[dst] += h[src] * dinv[src]*dinv[dst]; 32 threads/edge, 4 feats/thread
__global__ void edge_scatter(const int* __restrict__ src,
                             const int* __restrict__ dst,
                             const float* __restrict__ dinv,
                             const float* __restrict__ h,
                             float* __restrict__ agg, int e) {
    int t = blockIdx.x * blockDim.x + threadIdx.x;
    int ed = t >> 5;
    if (ed >= e) return;
    int f = (t & 31) << 2;
    int s = src[ed], d = dst[ed];
    float nrm = dinv[s] * dinv[d];
    const float4 hv = *(const float4*)(h + (size_t)s * WID + f);
    float* ap = agg + (size_t)d * WID + f;
    atomicAdd(ap + 0, hv.x * nrm);
    atomicAdd(ap + 1, hv.y * nrm);
    atomicAdd(ap + 2, hv.z * nrm);
    atomicAdd(ap + 3, hv.w * nrm);
}

// out = relu(agg + b)
__global__ void gcn_finalize(const float* __restrict__ agg,
                             const float* __restrict__ b,
                             float* __restrict__ out, int total) {
    int t = blockIdx.x * blockDim.x + threadIdx.x;
    if (t < total) out[t] = fmaxf(agg[t] + b[t & (WID - 1)], 0.0f);
}

// Per-column mean / rstd over n rows (block per column).
__global__ void bn_stats(const float* __restrict__ X, int n, int cols,
                         float* __restrict__ mu, float* __restrict__ rstd) {
    __shared__ float ss[256];
    __shared__ float sq[256];
    int c = blockIdx.x;
    float s = 0.0f, q = 0.0f;
    for (int r = threadIdx.x; r < n; r += blockDim.x) {
        float v = X[(size_t)r * cols + c];
        s += v;
        q += v * v;
    }
    ss[threadIdx.x] = s;
    sq[threadIdx.x] = q;
    __syncthreads();
    for (int o = blockDim.x >> 1; o > 0; o >>= 1) {
        if ((int)threadIdx.x < o) {
            ss[threadIdx.x] += ss[threadIdx.x + o];
            sq[threadIdx.x] += sq[threadIdx.x + o];
        }
        __syncthreads();
    }
    if (threadIdx.x == 0) {
        float m = ss[0] / (float)n;
        float v = sq[0] / (float)n - m * m;
        mu[c]   = m;
        rstd[c] = rsqrtf(v + 1e-5f);
    }
}

// y = relu((x - mu)*rstd*g + be)
__global__ void bn_apply(const float* __restrict__ X,
                         const float* __restrict__ mu,
                         const float* __restrict__ rstd,
                         const float* __restrict__ g,
                         const float* __restrict__ be,
                         float* __restrict__ Y, int total, int cols) {
    int t = blockIdx.x * blockDim.x + threadIdx.x;
    if (t < total) {
        int c = t % cols;
        float v = (X[t] - mu[c]) * rstd[c] * g[c] + be[c];
        Y[t] = fmaxf(v, 0.0f);
    }
}

// segment_max via uint atomicMax (values are post-ReLU, i.e. >= 0)
__global__ void pool_max(const float* __restrict__ X,
                         const int* __restrict__ batch,
                         float* __restrict__ pooled, int total) {
    int t = blockIdx.x * blockDim.x + threadIdx.x;
    if (t < total) {
        int node = t >> 7;
        int f = t & (WID - 1);
        int gph = batch[node];
        atomicMax((unsigned int*)&pooled[gph * WID + f],
                  __float_as_uint(X[t]));
    }
}

// out[g] = dot(t5n[g,:64], W6) + b6
__global__ void final_out(const float* __restrict__ t5n,
                          const float* __restrict__ W6,
                          const float* __restrict__ b6,
                          float* __restrict__ out) {
    int gph = threadIdx.x;
    if (gph < NGRAPH) {
        float s = 0.0f;
        for (int j = 0; j < 64; ++j) s += t5n[gph * 64 + j] * W6[j];
        out[gph] = s + b6[0];
    }
}

// ---------------------------------------------------------------------------
extern "C" void kernel_launch(void* const* d_in, const int* in_sizes, int n_in,
                              void* d_out, int out_size, void* d_ws, size_t ws_size,
                              hipStream_t stream) {
    const float* x     = (const float*)d_in[0];
    const int*   ei    = (const int*)d_in[1];
    const int*   batch = (const int*)d_in[2];
    const float* W1 = (const float*)d_in[3];
    const float* b1 = (const float*)d_in[4];
    const float* W2 = (const float*)d_in[5];
    const float* b2 = (const float*)d_in[6];
    const float* W3 = (const float*)d_in[7];
    const float* b3 = (const float*)d_in[8];
    const float* g1 = (const float*)d_in[9];
    const float* be1 = (const float*)d_in[10];
    const float* W4 = (const float*)d_in[11];
    const float* b4 = (const float*)d_in[12];
    const float* W5 = (const float*)d_in[13];
    const float* b5 = (const float*)d_in[14];
    const float* g2 = (const float*)d_in[15];
    const float* be2 = (const float*)d_in[16];
    const float* W6 = (const float*)d_in[17];
    const float* b6 = (const float*)d_in[18];
    float* out = (float*)d_out;

    const int N = NNODES, E = NEDGES;
    const int* src = ei;
    const int* dst = ei + E;

    // Workspace carve (all 16B aligned): 2 big N x 128 buffers + small state.
    char* w = (char*)d_ws;
    float* bufA  = (float*)w;                  w += (size_t)N * WID * 4;
    float* bufB  = (float*)w;                  w += (size_t)N * WID * 4;
    float* deg   = (float*)w;                  w += (size_t)N * 4;
    float* dinv  = (float*)w;                  w += (size_t)N * 4;
    float* mu    = (float*)w;                  w += 128 * 4;
    float* rstd  = (float*)w;                  w += 128 * 4;
    float* pooled = (float*)w;                 w += NGRAPH * WID * 4;
    float* t5    = (float*)w;                  w += NGRAPH * 64 * 4;
    float* t5n   = (float*)w;                  w += NGRAPH * 64 * 4;
    float* mu2   = (float*)w;                  w += 64 * 4;
    float* rstd2 = (float*)w;                  w += 64 * 4;

    const int total = N * WID;                 // 12.8M elements
    const int eltBlocks  = (total + 255) / 256;
    const int rowTiles   = N / 16;             // 6250 (N % 16 == 0)
    const int edgeBlocks = (E * 32 + 255) / 256;

    // --- degrees -----------------------------------------------------------
    zero_f<<<(N + 255) / 256, 256, 0, stream>>>(deg, N);
    deg_count<<<(E + 255) / 256, 256, 0, stream>>>(dst, E, deg);
    make_dinv<<<(N + 255) / 256, 256, 0, stream>>>(deg, dinv, N);

    // --- GCN layer 1:  h1 = x @ W1 ; agg ; relu(+b1) -----------------------
    wmma_gemm<FIN, WID, false, false><<<rowTiles, 256, 0, stream>>>(x, W1, nullptr, bufA);
    self_init<<<eltBlocks, 256, 0, stream>>>(bufA, dinv, bufB, total);
    edge_scatter<<<edgeBlocks, 256, 0, stream>>>(src, dst, dinv, bufA, bufB, E);
    gcn_finalize<<<eltBlocks, 256, 0, stream>>>(bufB, b1, bufA, total);

    // --- GCN layer 2 -------------------------------------------------------
    wmma_gemm<WID, WID, false, false><<<rowTiles, 256, 0, stream>>>(bufA, W2, nullptr, bufB);
    self_init<<<eltBlocks, 256, 0, stream>>>(bufB, dinv, bufA, total);
    edge_scatter<<<edgeBlocks, 256, 0, stream>>>(src, dst, dinv, bufB, bufA, E);
    gcn_finalize<<<eltBlocks, 256, 0, stream>>>(bufA, b2, bufB, total);

    // --- mlp_first: Linear(+b3) -> BN -> ReLU -> Linear(+b4) -> ReLU -------
    wmma_gemm<WID, WID, false, true><<<rowTiles, 256, 0, stream>>>(bufB, W3, b3, bufA);
    bn_stats<<<WID, 256, 0, stream>>>(bufA, N, WID, mu, rstd);
    bn_apply<<<eltBlocks, 256, 0, stream>>>(bufA, mu, rstd, g1, be1, bufB, total, WID);
    wmma_gemm<WID, WID, true, true><<<rowTiles, 256, 0, stream>>>(bufB, W4, b4, bufA);

    // --- global max pool ---------------------------------------------------
    zero_f<<<(NGRAPH * WID + 255) / 256, 256, 0, stream>>>(pooled, NGRAPH * WID);
    pool_max<<<eltBlocks, 256, 0, stream>>>(bufA, batch, pooled, total);

    // --- head: Linear(+b5) -> BN -> ReLU -> Linear(+b6) --------------------
    wmma_gemm<WID, 64, false, true><<<NGRAPH / 16, 128, 0, stream>>>(pooled, W5, b5, t5);
    bn_stats<<<64, 256, 0, stream>>>(t5, NGRAPH, 64, mu2, rstd2);
    bn_apply<<<(NGRAPH * 64 + 255) / 256, 256, 0, stream>>>(t5, mu2, rstd2, g2, be2, t5n, NGRAPH * 64, 64);
    final_out<<<1, 64, 0, stream>>>(t5n, W6, b6, out);

    (void)in_sizes; (void)n_in; (void)out_size; (void)ws_size;
}
